// GCNLayer_19439021981955
// MI455X (gfx1250) — compile-verified, matrix operand order checked by reading, not run on hardware
//
#include <hip/hip_runtime.h>

// CDNA5 WMMA types
typedef __attribute__((ext_vector_type(16))) __bf16 v16bf;
typedef __attribute__((ext_vector_type(8)))  float  v8f;

#define N_DIM 16384
#define C_DIM 128

// bf16 A/B fragment (16x32): lane<16 holds row M=lane, K in {0..7}u{16..23};
// lane>=16 holds row M=lane-16, K in {8..15}u{24..31}.  Two 16B chunks.
struct Frag32 { uint4 lo, hi; };
__device__ __forceinline__ v16bf load_frag(const __bf16* p) {
    Frag32 f;
    f.lo = *(const uint4*)(p);        // K base+0 .. base+7
    f.hi = *(const uint4*)(p + 16);   // K base+16 .. base+23
    return __builtin_bit_cast(v16bf, f);
}

// One v_cvt_pk_bf16_f32, kept opaque so clang cannot scalarize into
// v_mov_b16 half-shuffles (seen in previous rounds' codegen).
__device__ __forceinline__ unsigned int pack2(float a, float b) {
    unsigned int r;
    asm("v_cvt_pk_bf16_f32 %0, %1, %2" : "=v"(r) : "v"(a), "v"(b));
    return r;
}

// Build A fragment from 16 f32 as exactly 8 packed converts.
struct U8P { unsigned int u[8]; };
__device__ __forceinline__ v16bf cvt_frag(const float4* q) {
    U8P h;
    #pragma unroll
    for (int i = 0; i < 4; ++i) {
        h.u[2*i+0] = pack2(q[i].x, q[i].y);
        h.u[2*i+1] = pack2(q[i].z, q[i].w);
    }
    return __builtin_bit_cast(v16bf, h);
}

__device__ __forceinline__ void mm_step(v16bf a, const __bf16* bfrag, v8f* acc) {
    #pragma unroll
    for (int j = 0; j < 8; ++j) {
        v16bf b = load_frag(bfrag + j * 640);   // j*16 rows * stride 40
        acc[j] = __builtin_amdgcn_wmma_f32_16x16x32_bf16(
            false, a, false, b, (short)0, acc[j], false, false);
    }
}

__global__ __launch_bounds__(128)
void gcn_fused_kernel(const float* __restrict__ x,
                      const float* __restrict__ x0,
                      const float* __restrict__ adj,
                      const float* __restrict__ w1,
                      float* __restrict__ out)
{
    // Phase-1 view: Bbuf[2] : 2 x [128][40] bf16 (2 x 10240B), double buffered
    // Phase-2 view: S [64][136] bf16 (17408B) + Wt [128][136] bf16 (34816B)
    __shared__ __align__(16) unsigned char smem[52224];
    __bf16* Bbuf0 = (__bf16*)(smem);
    __bf16* Bbuf1 = (__bf16*)(smem + 10240);
    __bf16* S     = (__bf16*)(smem);
    __bf16* Wt    = (__bf16*)(smem + 17408);

    const int tid  = threadIdx.x;
    const int wave = tid >> 5;
    const int lane = tid & 31;
    const int l15  = lane & 15;
    const int hi8  = (lane < 16) ? 0 : 8;
    const int kb   = hi8;                 // per-lane K base inside fragment
    const int wm   = wave * 16;           // wave's 16-row strip (unique per wave)
    const int m0   = blockIdx.x * 64;

    // ---- A: fragments straight from global (adj read exactly once) --------
    const float* arow = adj + (size_t)(m0 + wm + l15) * N_DIM + kb;

    // ---- B staging bases: blk = tid + 128p => cp = tid&63 (p-invariant),
    //      k = 2*(tid>>6) + 4p  => everything is base + immediate offset ----
    const int cp = tid & 63, qk = tid >> 6;
    const float*  bgbase = x + (size_t)(2 * qk) * C_DIM + 2 * cp;   // + ks*C + p*4*C
    __bf16* sb0 = Bbuf0 + (2 * cp) * 40 + 2 * qk;                   // + p*4 (+40)
    __bf16* sb1 = Bbuf1 + (2 * cp) * 40 + 2 * qk;
    const __bf16* bf0 = Bbuf0 + l15 * 40 + kb;                      // frag bases
    const __bf16* bf1 = Bbuf1 + l15 * 40 + kb;

    v8f acc[8] = {};
    float4 va[4];            // pipelined A regs (one K-step ahead)
    float2 vba[8], vbb[8];   // pipelined B regs (2x2 sub-blocks)

    auto loadA = [&](int ks) {
        const float* ap = arow + ks;
        va[0] = *(const float4*)(ap);        // K +0..3
        va[1] = *(const float4*)(ap + 4);    // K +4..7
        va[2] = *(const float4*)(ap + 16);   // K +16..19
        va[3] = *(const float4*)(ap + 20);   // K +20..23
        __builtin_prefetch(ap + 64, 0, 1);   // 2 K-steps ahead -> global_prefetch_b8
    };
    auto loadB = [&](int ks) {
        const float* bp = bgbase + (size_t)ks * C_DIM;
        #pragma unroll
        for (int p = 0; p < 8; ++p) {
            vba[p] = *(const float2*)(bp + p * 4 * C_DIM);           // row k
            vbb[p] = *(const float2*)(bp + p * 4 * C_DIM + C_DIM);   // row k+1
        }
    };
    auto storeB = [&](__bf16* sb) {          // Bs[n][k] transposed, stride 40
        #pragma unroll
        for (int p = 0; p < 8; ++p) {
            *(unsigned int*)(sb + p * 4)      = pack2(vba[p].x, vbb[p].x);  // col c
            *(unsigned int*)(sb + p * 4 + 40) = pack2(vba[p].y, vbb[p].y);  // col c+1
        }
    };

    // ---------------- Phase 1: h = adj @ x  (K = 16384, KT = 32) -----------
    loadA(0);
    loadB(0);
    storeB(sb0);
    __syncthreads();

    // two K-steps per iteration -> fixed buffer roles, no select instructions
    for (int ks = 0; ks < N_DIM - 64; ks += 64) {
        {   v16bf a = cvt_frag(va);
            loadA(ks + 32); loadB(ks + 32);
            mm_step(a, bf0, acc);            // compute tile ks from buf0
            storeB(sb1);                     // stage tile ks+32 into buf1
            __syncthreads();
        }
        {   v16bf a = cvt_frag(va);
            loadA(ks + 64); loadB(ks + 64);
            mm_step(a, bf1, acc);            // compute tile ks+32 from buf1
            storeB(sb0);                     // stage tile ks+64 into buf0
            __syncthreads();
        }
    }
    {   // last two K-steps (N_DIM-64, N_DIM-32)
        v16bf a = cvt_frag(va);
        loadA(N_DIM - 32); loadB(N_DIM - 32);
        mm_step(a, bf0, acc);
        storeB(sb1);
        __syncthreads();
        a = cvt_frag(va);
        mm_step(a, bf1, acc);
    }

    // ---------------- alpha blend: out1 = 0.9*h + 0.1*x_0 ------------------
    const int mrow = m0 + wm + hi8;   // C/D layout: VGPR r -> row r (+8 hi lanes)
    const float* x0p = x0 + (size_t)mrow * C_DIM + l15;
    #pragma unroll
    for (int j = 0; j < 8; ++j)
        #pragma unroll
        for (int r = 0; r < 8; ++r)
            acc[j][r] = 0.9f * acc[j][r] + 0.1f * x0p[r * C_DIM + j * 16];
    __syncthreads();   // all waves done with Bbuf before S overwrites it

    // ------------- Phase 2: out = (1-B)*out1 + B*(out1 @ W) ----------------
    {   // stage out1 as bf16 in A-source order, S[64][136]
        __bf16* Sp = S + (wm + hi8) * 136 + l15;
        #pragma unroll
        for (int j = 0; j < 8; ++j)
            #pragma unroll
            for (int r = 0; r < 8; ++r)
                Sp[r * 136 + j * 16] = (__bf16)acc[j][r];
    }
    #pragma unroll
    for (int p = 0; p < 32; ++p) {    // Wt[n][k] = W[k][n]
        int idx = tid + p * 128;
        int k = idx >> 5, c4 = (idx & 31) << 2;
        float4 v = *(const float4*)(w1 + k * C_DIM + c4);
        float vv[4] = { v.x, v.y, v.z, v.w };
        #pragma unroll
        for (int i = 0; i < 4; ++i)
            Wt[(c4 + i) * 136 + k] = (__bf16)vv[i];
    }
    __syncthreads();

    v8f acc2[8] = {};
    {
        const __bf16* ap2 = S + (wm + l15) * 136 + kb;
        const __bf16* bp2 = Wt + l15 * 136 + kb;
        #pragma unroll
        for (int kk = 0; kk < 128; kk += 32) {
            v16bf a2 = load_frag(ap2 + kk);
            #pragma unroll
            for (int j = 0; j < 8; ++j) {
                v16bf b2 = load_frag(bp2 + j * 2176 + kk);   // j*16 rows * 136
                acc2[j] = __builtin_amdgcn_wmma_f32_16x16x32_bf16(
                    false, a2, false, b2, (short)0, acc2[j], false, false);
            }
        }
    }

    const float BETA = 0.11778303565638346f;   // log(1.125)
    const float OMB  = 1.0f - BETA;
    float* outp = out + (size_t)mrow * C_DIM + l15;
    #pragma unroll
    for (int j = 0; j < 8; ++j)
        #pragma unroll
        for (int r = 0; r < 8; ++r)
            outp[r * C_DIM + j * 16] = OMB * acc[j][r] + BETA * acc2[j][r];
}

extern "C" void kernel_launch(void* const* d_in, const int* in_sizes, int n_in,
                              void* d_out, int out_size, void* d_ws, size_t ws_size,
                              hipStream_t stream) {
    const float* x   = (const float*)d_in[0];   // [16384,128]
    const float* x0  = (const float*)d_in[1];   // [16384,128]
    const float* adj = (const float*)d_in[2];   // [16384,16384]
    const float* w1  = (const float*)d_in[3];   // [128,128]
    float* out = (float*)d_out;                 // [16384,128]
    (void)in_sizes; (void)n_in; (void)out_size; (void)d_ws; (void)ws_size;

    gcn_fused_kernel<<<dim3(N_DIM / 64), dim3(128), 0, stream>>>(x, x0, adj, w1, out);
}
